// GeneralMambaBlock_76493367542166
// MI455X (gfx1250) — matrix operand order repaired and checked by reading, not compiled, not run on hardware
//
#include <hip/hip_runtime.h>
#include <hip/hip_bf16.h>

// ---------------------------------------------------------------------------
// Mamba block for MI455X (gfx1250): WMMA f16 GEMMs with f16-resident operands,
// state-parallel selective scan (16 lanes per (b,d), shuffle reduction).
// Shapes: B=2, L=2048, D=1024, DI=2048, DS=16, DC=4, DTR=64. M = B*L = 4096.
// ---------------------------------------------------------------------------

typedef __attribute__((ext_vector_type(16))) _Float16 v16h;
typedef __attribute__((ext_vector_type(8)))  _Float16 v8h;
typedef __attribute__((ext_vector_type(8)))  float    v8f;

#define M_ROWS 4096
#define DMODEL 1024
#define DINNER 2048
#define LSEQ   2048

// ---- WMMA fragment helpers (ISA 7.12.2 layouts, f16-native loads) ---------

__device__ __forceinline__ v16h load_a_frag16(const _Float16* __restrict__ A, int lda,
                                              int m0, int k0, int lane) {
  const int half = lane >> 4;        // 0: K 0-7 & 16-23, 1: K 8-15 & 24-31
  const int r    = lane & 15;        // row within 16x32 tile
  const _Float16* p = A + (size_t)(m0 + r) * lda + k0 + half * 8;
  union { v16h v; v8h h[2]; } u;
  u.h[0] = *(const v8h*)(p);         // K half*8 .. half*8+7
  u.h[1] = *(const v8h*)(p + 16);    // K 16+half*8 .. 16+half*8+7
  return u.v;
}

// B tile (32x16) from row-major W (N x K): B[k][n] = W[n0+n][k0+k]
__device__ __forceinline__ v16h load_b_frag16(const _Float16* __restrict__ W, int ldw,
                                              int n0, int k0, int lane) {
  const int half = lane >> 4;        // 0: K 0-7 & 16-23, 1: K 8-15 & 24-31
  const int c    = lane & 15;        // column within tile
  const _Float16* p = W + (size_t)(n0 + c) * ldw + k0 + half * 8;
  union { v16h v; v8h h[2]; } u;
  u.h[0] = *(const v8h*)(p);
  u.h[1] = *(const v8h*)(p + 16);
  return u.v;
}

__device__ __forceinline__ void store_c_frag(float* __restrict__ C, int ldc,
                                             int m0, int n0, int lane, v8f c) {
  const int half = lane >> 4;
  const int col  = lane & 15;
#pragma unroll
  for (int r = 0; r < 8; ++r)
    C[(size_t)(m0 + r + half * 8) * ldc + n0 + col] = c[r];
}

// ---- weight f32 -> f16 conversion ----------------------------------------

__global__ __launch_bounds__(256) void cvt_f16_kernel(const float* __restrict__ src,
                                                      _Float16* __restrict__ dst, int n) {
  const int i = blockIdx.x * 256 + threadIdx.x;
  if (i < n) dst[i] = (_Float16)src[i];
}

// ---- K1: fused residual-add + LayerNorm (h emitted as f16) ----------------

__global__ __launch_bounds__(256) void addnorm_kernel(
    const float* __restrict__ x, const float* __restrict__ res_in,
    const float* __restrict__ ln_w, const float* __restrict__ ln_b,
    float* __restrict__ res_out, _Float16* __restrict__ h16) {
  const int row = blockIdx.x;          // 0..4095
  const int tid = threadIdx.x;
  const float* xr = x      + (size_t)row * DMODEL;
  const float* rr = res_in + (size_t)row * DMODEL;
  float v[4], s = 0.f, s2 = 0.f;
#pragma unroll
  for (int j = 0; j < 4; ++j) {
    const int c = tid + 256 * j;
    const float t = xr[c] + rr[c];
    v[j] = t; s += t; s2 += t * t;
    res_out[(size_t)row * DMODEL + c] = t;
  }
#pragma unroll
  for (int off = 16; off; off >>= 1) {
    s  += __shfl_xor(s,  off, 32);
    s2 += __shfl_xor(s2, off, 32);
  }
  __shared__ float ss[8], ss2[8];
  const int wave = tid >> 5, lane = tid & 31;
  if (lane == 0) { ss[wave] = s; ss2[wave] = s2; }
  __syncthreads();
  if (tid == 0) {
    float a = 0.f, b = 0.f;
#pragma unroll
    for (int i = 0; i < 8; ++i) { a += ss[i]; b += ss2[i]; }
    ss[0] = a; ss2[0] = b;
  }
  __syncthreads();
  const float mean = ss[0] * (1.f / DMODEL);
  const float var  = ss2[0] * (1.f / DMODEL) - mean * mean;
  const float rstd = rsqrtf(var + 1e-5f);
#pragma unroll
  for (int j = 0; j < 4; ++j) {
    const int c = tid + 256 * j;
    h16[(size_t)row * DMODEL + c] = (_Float16)((v[j] - mean) * rstd * ln_w[c] + ln_b[c]);
  }
}

// ---- Main WMMA GEMM: C[MxN] = A[MxK] * W[NxK]^T (A, W in f16) -------------
// Block = 8 waves (2x4), each wave computes 64x64 (4x4 tiles of 16x16).
// EPI: 0 = plain f32 store, 1 = softplus(c + bias[n]).

template <int EPI>
__global__ __launch_bounds__(256) void gemm_wmma_kernel(
    const _Float16* __restrict__ A, int lda,
    const _Float16* __restrict__ W, int ldw,
    float* __restrict__ C, int ldc, int K,
    const float* __restrict__ bias) {
  const int tid  = threadIdx.x;
  const int lane = tid & 31;
  const int wave = tid >> 5;                 // 0..7
  const int m_base = blockIdx.y * 128 + (wave >> 2) * 64;
  const int n_base = blockIdx.x * 256 + (wave & 3) * 64;

  v8f acc[4][4];
#pragma unroll
  for (int i = 0; i < 4; ++i)
#pragma unroll
    for (int j = 0; j < 4; ++j) acc[i][j] = (v8f){0.f,0.f,0.f,0.f,0.f,0.f,0.f,0.f};

  for (int k0 = 0; k0 < K; k0 += 32) {
    v16h af[4], bfr[4];
#pragma unroll
    for (int i = 0; i < 4; ++i) af[i]  = load_a_frag16(A, lda, m_base + 16 * i, k0, lane);
#pragma unroll
    for (int j = 0; j < 4; ++j) bfr[j] = load_b_frag16(W, ldw, n_base + 16 * j, k0, lane);
#pragma unroll
    for (int i = 0; i < 4; ++i)
#pragma unroll
      for (int j = 0; j < 4; ++j)
        acc[i][j] = __builtin_amdgcn_wmma_f32_16x16x32_f16(
            false, af[i], false, bfr[j], (short)0, acc[i][j], false, false);
  }

#pragma unroll
  for (int i = 0; i < 4; ++i)
#pragma unroll
    for (int j = 0; j < 4; ++j) {
      v8f c = acc[i][j];
      if (EPI == 1) {
        const int col = n_base + 16 * j + (lane & 15);
        const float bb = bias[col];
#pragma unroll
        for (int r = 0; r < 8; ++r) {
          const float t = c[r] + bb;
          c[r] = (t > 20.f) ? t : log1pf(expf(t));   // softplus
        }
      }
      store_c_frag(C, ldc, m_base + 16 * i, n_base + 16 * j, lane, c);
    }
}

// ---- Narrow-N WMMA GEMM (N=96): wave owns a 16x96 strip -------------------
// Epilogue: cols 0-63 (dt input) stored f16 for the dt GEMM; cols 64-95
// (Bm|Cm) stored f32 for the scan.

__global__ __launch_bounds__(256) void gemm_wmma_n96_kernel(
    const _Float16* __restrict__ A, int lda,
    const _Float16* __restrict__ W, int ldw,
    float* __restrict__ Cf, _Float16* __restrict__ Ch, int K) {
  const int tid  = threadIdx.x;
  const int lane = tid & 31;
  const int wave = tid >> 5;
  const int m0 = (blockIdx.x * 8 + wave) * 16;
  const int half = lane >> 4;
  const int col  = lane & 15;

  v8f acc[6];
#pragma unroll
  for (int j = 0; j < 6; ++j) acc[j] = (v8f){0.f,0.f,0.f,0.f,0.f,0.f,0.f,0.f};

  for (int k0 = 0; k0 < K; k0 += 32) {
    const v16h a = load_a_frag16(A, lda, m0, k0, lane);
#pragma unroll
    for (int j = 0; j < 6; ++j) {
      const v16h b = load_b_frag16(W, ldw, 16 * j, k0, lane);
      acc[j] = __builtin_amdgcn_wmma_f32_16x16x32_f16(
          false, a, false, b, (short)0, acc[j], false, false);
    }
  }
#pragma unroll
  for (int j = 0; j < 4; ++j)        // dt part -> f16 (M x 64)
#pragma unroll
    for (int r = 0; r < 8; ++r)
      Ch[(size_t)(m0 + r + half * 8) * 64 + 16 * j + col] = (_Float16)acc[j][r];
#pragma unroll
  for (int j = 4; j < 6; ++j)        // Bm|Cm part -> f32 (M x 96 layout)
#pragma unroll
    for (int r = 0; r < 8; ++r)
      Cf[(size_t)(m0 + r + half * 8) * 96 + 16 * j + col] = acc[j][r];
}

// ---- K3: causal depthwise conv (DC=4) + SiLU (f32 + f16 outputs) ----------

__global__ __launch_bounds__(256) void conv_silu_kernel(
    const float* __restrict__ xz, const float* __restrict__ conv_w,
    const float* __restrict__ conv_b, float* __restrict__ xc,
    _Float16* __restrict__ xc16) {
  const size_t idx = (size_t)blockIdx.x * 256 + threadIdx.x; // m*DI + c
  const int c = (int)(idx & (DINNER - 1));
  const int m = (int)(idx >> 11);
  const int l = m & (LSEQ - 1);
  const int b = m >> 11;
  float s = conv_b[c];
#pragma unroll
  for (int j = 0; j < 4; ++j) {
    const int ll = l - 3 + j;
    if (ll >= 0)
      s += xz[(size_t)(b * LSEQ + ll) * (2 * DINNER) + c] * conv_w[c * 4 + j];
  }
  const float r = s / (1.f + __expf(-s));    // SiLU
  xc[idx] = r;
  xc16[idx] = (_Float16)r;
}

// ---- K6: state-parallel selective scan ------------------------------------
// 16 lanes per (b,d) scan: lane owns one SSM state s. 65536 threads = 2048
// waves. Loop-carried chain per step: v_exp_f32 + fma. y = sum_s h*C via
// __shfl_xor reduction inside the aligned 16-lane group (off-critical-path).
// Fuses D*x skip and SiLU(z) gating; emits y as f16 for the out-proj GEMM.

__global__ __launch_bounds__(256) void scan_kernel(
    const float* __restrict__ xdbl,   // (M, 96) : cols 64..95 = [Bm(16)|Cm(16)]
    const float* __restrict__ dtf,    // (M, DI) softplus'd dt
    const float* __restrict__ xc,     // (M, DI)
    const float* __restrict__ xz,     // (M, 2*DI), z = cols [DI, 2*DI)
    const float* __restrict__ A_log,  // (DI, 16)
    const float* __restrict__ D_param,
    _Float16* __restrict__ y16) {     // (M, DI) f16
  const int gid  = blockIdx.x * 256 + threadIdx.x;   // 0 .. 65535
  const int s    = gid & 15;                          // state index
  const int pair = gid >> 4;                          // 0 .. 4095 = (b,d)
  const int b    = pair >> 11;
  const int d    = pair & (DINNER - 1);

  const float Ar = -__expf(A_log[d * 16 + s]);
  const float Dp = D_param[d];
  float hs = 0.f;

  for (int l = 0; l < LSEQ; ++l) {
    const int m = b * LSEQ + l;
    const float dt = dtf[(size_t)m * DINNER + d];     // broadcast in 16-lane grp
    const float xv = xc [(size_t)m * DINNER + d];     // broadcast
    const float Bs = xdbl[(size_t)m * 96 + 64 + s];   // coalesced 16 floats
    const float Cs = xdbl[(size_t)m * 96 + 80 + s];   // coalesced 16 floats

    hs = __expf(dt * Ar) * hs + (dt * xv) * Bs;       // carried chain
    float part = hs * Cs;
#pragma unroll
    for (int off = 8; off; off >>= 1)                 // stays in 16-lane group
      part += __shfl_xor(part, off, 32);

    if (s == 0) {
      const float z = xz[(size_t)m * (2 * DINNER) + DINNER + d];
      const float silu_z = z / (1.f + __expf(-z));
      y16[(size_t)m * DINNER + d] = (_Float16)((part + Dp * xv) * silu_z);
    }
  }
}

// ---------------------------------------------------------------------------

extern "C" void kernel_launch(void* const* d_in, const int* in_sizes, int n_in,
                              void* d_out, int out_size, void* d_ws, size_t ws_size,
                              hipStream_t stream) {
  const float* x         = (const float*)d_in[0];
  const float* residual  = (const float*)d_in[1];
  const float* ln_w      = (const float*)d_in[2];
  const float* ln_b      = (const float*)d_in[3];
  const float* in_proj_w = (const float*)d_in[4];   // (4096, 1024)
  const float* conv_w    = (const float*)d_in[5];   // (2048, 4)
  const float* conv_b    = (const float*)d_in[6];
  const float* x_proj_w  = (const float*)d_in[7];   // (96, 2048)
  const float* dt_proj_w = (const float*)d_in[8];   // (2048, 64)
  const float* dt_proj_b = (const float*)d_in[9];
  const float* A_log     = (const float*)d_in[10];  // (2048, 16)
  const float* D_param   = (const float*)d_in[11];
  const float* out_proj_w= (const float*)d_in[12];  // (1024, 2048)

  float* out     = (float*)d_out;                    // (B,L,D)
  float* res_out = out + (size_t)M_ROWS * DMODEL;    // second tuple element

  // ---- workspace layout ----
  float* xz   = (float*)d_ws;                        // 4096*4096 f32
  float* dtf  = xz   + (size_t)M_ROWS * 2 * DINNER;  // 4096*2048 f32
  float* xc   = dtf  + (size_t)M_ROWS * DINNER;      // 4096*2048 f32
  float* xdbl = xc   + (size_t)M_ROWS * DINNER;      // 4096*96   f32
  _Float16* h16    = (_Float16*)(xdbl + (size_t)M_ROWS * 96);
  _Float16* xc16   = h16    + (size_t)M_ROWS * DMODEL;   // 4096*2048
  _Float16* y16    = xc16   + (size_t)M_ROWS * DINNER;   // 4096*2048
  _Float16* xdbl16 = y16    + (size_t)M_ROWS * DINNER;   // 4096*64
  _Float16* wi16   = xdbl16 + (size_t)M_ROWS * 64;       // 4096*1024
  _Float16* wx16   = wi16   + (size_t)4096 * 1024;       // 96*2048
  _Float16* wdt16  = wx16   + (size_t)96 * 2048;         // 2048*64
  _Float16* wo16   = wdt16  + (size_t)2048 * 64;         // 1024*2048

  // 0. weights f32 -> f16 (L2-resident afterwards)
  cvt_f16_kernel<<<(4096 * 1024) / 256, 256, 0, stream>>>(in_proj_w, wi16, 4096 * 1024);
  cvt_f16_kernel<<<(96 * 2048) / 256, 256, 0, stream>>>(x_proj_w, wx16, 96 * 2048);
  cvt_f16_kernel<<<(2048 * 64) / 256, 256, 0, stream>>>(dt_proj_w, wdt16, 2048 * 64);
  cvt_f16_kernel<<<(1024 * 2048) / 256, 256, 0, stream>>>(out_proj_w, wo16, 1024 * 2048);

  // 1. residual-add + LayerNorm
  addnorm_kernel<<<M_ROWS, 256, 0, stream>>>(x, residual, ln_w, ln_b, res_out, h16);

  // 2. xz = h @ in_proj_w^T   (4096x1024 x 1024x4096)
  gemm_wmma_kernel<0><<<dim3(16, 32), 256, 0, stream>>>(
      h16, DMODEL, wi16, DMODEL, xz, 2 * DINNER, DMODEL, nullptr);

  // 3. causal depthwise conv + SiLU on x_in half of xz
  conv_silu_kernel<<<(M_ROWS * DINNER) / 256, 256, 0, stream>>>(xz, conv_w, conv_b, xc, xc16);

  // 4. x_dbl = xc @ x_proj_w^T   (4096x2048 x 2048x96)
  gemm_wmma_n96_kernel<<<32, 256, 0, stream>>>(xc16, DINNER, wx16, DINNER, xdbl, xdbl16, DINNER);

  // 5. dt = softplus(x_dbl[:, :64] @ dt_proj_w^T + dt_bias)   (K=64, N=2048)
  gemm_wmma_kernel<1><<<dim3(8, 32), 256, 0, stream>>>(
      xdbl16, 64, wdt16, 64, dtf, DINNER, 64, dt_proj_b);

  // 6. state-parallel selective scan + D*x skip + SiLU(z) gating -> y (f16)
  scan_kernel<<<256, 256, 0, stream>>>(xdbl, dtf, xc, xz, A_log, D_param, y16);

  // 7. out = y @ out_proj_w^T   (4096x2048 x 2048x1024)
  gemm_wmma_kernel<0><<<dim3(4, 32), 256, 0, stream>>>(
      y16, DINNER, wo16, DINNER, out, DMODEL, DINNER, nullptr);
}
